// VNEGNN_31928786878567
// MI455X (gfx1250) — matrix operand order repaired and checked by reading, not compiled, not run on hardware
//
#include <hip/hip_runtime.h>
#include <hip/hip_bf16.h>

typedef __bf16 bf16_t;
typedef __attribute__((ext_vector_type(16))) __bf16 v16bf;
typedef __attribute__((ext_vector_type(8)))  float  v8f;

#define HDIM 128
#define NCH  3
#define TILE 64

__device__ __forceinline__ float silu_f(float x) { return x / (1.0f + __expf(-x)); }

// ---------------------------------------------------------------------------
// A-fragment loader: 16-bit A matrix 16x32 per wave, from a row-major bf16
// LDS tile. K order used here (contiguous; matches the B pre-swizzle):
//   lanes 0-15 : K = {0..15}, lanes 16-31 : K = {16..31}
//   element e -> K = (lane/16)*16 + e
// ---------------------------------------------------------------------------
__device__ __forceinline__ v16bf load_a_frag(const bf16_t* __restrict__ tile,
                                             int lda, int row0, int k0, int lane) {
  const int g = lane >> 4;
  const int m = lane & 15;
  const bf16_t* rp = tile + (size_t)(row0 + m) * lda + k0;
  v16bf a;
#pragma unroll
  for (int e = 0; e < 16; ++e) {
    const int v = e >> 1, hh = e & 1;
    const int k = g * 16 + (v << 1) + hh;
    a[e] = rp[k];
  }
  return a;
}

// ---------------------------------------------------------------------------
// 64x128 GEMM step: OUT = act(A[64 x K] @ B + bias). 8 waves: wave w owns rows
// [16*(w/2), +16) and col tiles {4*(w%2) .. +3}. B fragments are pre-swizzled
// in global memory so each lane loads 16 contiguous bf16.
// Optionally atomically accumulates f32 results into gAcc[gBase[row] + n].
// ---------------------------------------------------------------------------
__device__ __forceinline__ void gemm_tile64(
    const bf16_t* __restrict__ A, int lda,
    const v16bf* __restrict__ Bf, int kchunks,
    const float* __restrict__ bias, int act,
    bf16_t* __restrict__ outLds, int ldo,
    float* __restrict__ gAcc, const int* __restrict__ gBase, float gScale,
    int wave, int lane) {
  const int rt = wave >> 1;
  const int cg = wave & 1;
  const int g  = lane >> 4;
  const int n0 = lane & 15;
#pragma unroll
  for (int ntl = 0; ntl < 4; ++ntl) {
    const int nt = cg * 4 + ntl;
    v8f acc = {};
    for (int kk = 0; kk < kchunks; ++kk) {
      v16bf a = load_a_frag(A, lda, rt * 16, kk * 32, lane);
      v16bf b = Bf[(kk * 8 + nt) * 32 + lane];
      acc = __builtin_amdgcn_wmma_f32_16x16x32_bf16(false, a, false, b,
                                                    (short)0, acc, false, false);
    }
    const int n = nt * 16 + n0;
    const float bv = bias ? bias[n] : 0.0f;
#pragma unroll
    for (int r = 0; r < 8; ++r) {
      const int row = rt * 16 + r + 8 * g;   // C/D layout: M = r + 8*(lane/16), N = lane%16
      float v = acc[r] + bv;
      if (act) v = silu_f(v);
      if (outLds) outLds[row * ldo + n] = (bf16_t)v;
      if (gAcc)   atomicAdd(&gAcc[(size_t)gBase[row] + n], v * gScale);
    }
  }
}

// 64-row x 1-col (padded to 16) matvec for the coord MLP second linear.
__device__ __forceinline__ void gemm_colvec64(
    const bf16_t* __restrict__ A, int lda,
    const v16bf* __restrict__ Bf, int kchunks,
    float* __restrict__ cmOut, int wave, int lane) {
  if (wave >= 4) return;
  const int rt = wave;
  const int g  = lane >> 4;
  v8f acc = {};
  for (int kk = 0; kk < kchunks; ++kk) {
    v16bf a = load_a_frag(A, lda, rt * 16, kk * 32, lane);
    v16bf b = Bf[kk * 32 + lane];
    acc = __builtin_amdgcn_wmma_f32_16x16x32_bf16(false, a, false, b,
                                                  (short)0, acc, false, false);
  }
  if ((lane & 15) == 0) {
#pragma unroll
    for (int r = 0; r < 8; ++r) cmOut[rt * 16 + r + 8 * g] = acc[r];
  }
}

// Final GEMM of a node MLP: out[row,n] = resid[row,n] + A@B + bias (no act),
// written straight to global f32.
__device__ __forceinline__ void gemm_tile64_out(
    const bf16_t* __restrict__ A, int lda,
    const v16bf* __restrict__ Bf, int kchunks,
    const float* __restrict__ bias, const float* __restrict__ resid,
    float* __restrict__ out, int row0, int wave, int lane) {
  const int rt = wave >> 1;
  const int cg = wave & 1;
  const int g  = lane >> 4;
  const int n0 = lane & 15;
#pragma unroll
  for (int ntl = 0; ntl < 4; ++ntl) {
    const int nt = cg * 4 + ntl;
    v8f acc = {};
    for (int kk = 0; kk < kchunks; ++kk) {
      v16bf a = load_a_frag(A, lda, rt * 16, kk * 32, lane);
      v16bf b = Bf[(kk * 8 + nt) * 32 + lane];
      acc = __builtin_amdgcn_wmma_f32_16x16x32_bf16(false, a, false, b,
                                                    (short)0, acc, false, false);
    }
    const int n = nt * 16 + n0;
    const float bv = bias[n];
#pragma unroll
    for (int r = 0; r < 8; ++r) {
      const int row = rt * 16 + r + 8 * g;
      const size_t gi = (size_t)(row0 + row) * HDIM + n;
      out[gi] = resid[gi] + acc[r] + bv;
    }
  }
}

// ---------------------------------------------------------------------------
// Weight pre-swizzle: w[K x N] f32 -> bf16 B-fragments laid out so that
// frag[((kk*NT + nt)*32 + lane)*16 + e] holds B[kk*32 + (lane/16)*16 + e][nt*16 + lane%16].
// Zero-pads K -> Kpad (mult of 32) and N -> Npad (mult of 16).
// ---------------------------------------------------------------------------
__global__ void prep_frag_kernel(const float* __restrict__ w, bf16_t* __restrict__ frag,
                                 int K, int Kpad, int N, int Npad) {
  const int idx = blockIdx.x * blockDim.x + threadIdx.x;
  const int NT = Npad >> 4;
  const int total = (Kpad >> 5) * NT * 512;
  if (idx >= total) return;
  const int e    = idx & 15;
  const int lane = (idx >> 4) & 31;
  const int fi   = idx >> 9;
  const int kk = fi / NT, nt = fi % NT;
  const int g = lane >> 4;
  const int n = nt * 16 + (lane & 15);
  const int k = kk * 32 + g * 16 + e;
  const float v = (k < K && n < N) ? w[(size_t)k * N + n] : 0.0f;
  frag[idx] = (bf16_t)v;
}

// ---------------------------------------------------------------------------
// A2A edge kernel: per 64-edge tile, gather [h[row] | h[col] | radial] -> bf16
// LDS, run edge MLP (silu o silu) + coord MLP via WMMA, atomically scatter:
//   agg[row] += ef ; cnum[row] += cd*cm ; cden[row] += 1
// ---------------------------------------------------------------------------
__global__ void a2a_edge_kernel(
    const float* __restrict__ h, const float* __restrict__ coord,
    const int* __restrict__ ei, int nE,
    const v16bf* __restrict__ w1f, const float* __restrict__ b1,
    const v16bf* __restrict__ w2f, const float* __restrict__ b2,
    const v16bf* __restrict__ cw1f, const float* __restrict__ cb1,
    const v16bf* __restrict__ cw2f,
    float* __restrict__ agg, float* __restrict__ cnum, float* __restrict__ cden) {
  __shared__ bf16_t A[TILE * 288];
  __shared__ bf16_t Mid[TILE * HDIM];
  __shared__ float sCd[TILE * 3];
  __shared__ float sCm[TILE];
  __shared__ int   sRow[TILE];
  __shared__ int   sBase[TILE];

  const int t = threadIdx.x, wave = t >> 5, lane = t & 31;
  const int r = t >> 2, s = t & 3;          // 4 threads per edge slot
  const int e = blockIdx.x * TILE + r;      // E is a multiple of TILE
  const int row = ei[e];
  const int col = ei[nE + e];
  if (s == 0) { sRow[r] = row; sBase[r] = row * HDIM; }

  const float4* hr4 = (const float4*)(h + (size_t)row * HDIM);
  const float4* hc4 = (const float4*)(h + (size_t)col * HDIM);
  __builtin_prefetch(hr4, 0, 0);
  __builtin_prefetch(hc4, 0, 0);
  for (int q = s * 8; q < s * 8 + 8; ++q) {
    float4 a = hr4[q], b = hc4[q];
    A[r * 288 + q * 4 + 0] = (bf16_t)a.x; A[r * 288 + q * 4 + 1] = (bf16_t)a.y;
    A[r * 288 + q * 4 + 2] = (bf16_t)a.z; A[r * 288 + q * 4 + 3] = (bf16_t)a.w;
    A[r * 288 + 128 + q * 4 + 0] = (bf16_t)b.x; A[r * 288 + 128 + q * 4 + 1] = (bf16_t)b.y;
    A[r * 288 + 128 + q * 4 + 2] = (bf16_t)b.z; A[r * 288 + 128 + q * 4 + 3] = (bf16_t)b.w;
  }
  if (s == 0) {
    float rad = 0.0f;
#pragma unroll
    for (int k = 0; k < 3; ++k) {
      float d = coord[row * 3 + k] - coord[col * 3 + k];
      sCd[r * 3 + k] = d; rad += d * d;
    }
    A[r * 288 + 256] = (bf16_t)rad;
    for (int c = 257; c < 288; ++c) A[r * 288 + c] = (bf16_t)0.0f;
  }
  __syncthreads();

  // edge MLP layer 1: [64 x 288] @ [288 x 128] -> silu -> Mid
  gemm_tile64(A, 288, w1f, 9, b1, 1, Mid, HDIM, nullptr, nullptr, 0.0f, wave, lane);
  __syncthreads();
  // edge MLP layer 2 -> silu -> ef (reuse A as [64 x 128]) + atomic agg[row] += ef
  gemm_tile64(Mid, HDIM, w2f, 4, b2, 1, A, HDIM, agg, sBase, 1.0f, wave, lane);
  __syncthreads();
  // coord MLP hidden = silu(ef @ cw1 + cb1)
  gemm_tile64(A, HDIM, cw1f, 4, cb1, 1, Mid, HDIM, nullptr, nullptr, 0.0f, wave, lane);
  __syncthreads();
  // cm = hidden @ cw2
  gemm_colvec64(Mid, HDIM, cw2f, 4, sCm, wave, lane);
  __syncthreads();

  if (t < TILE) {
    const float cm = sCm[t];
    const int rw = sRow[t];
#pragma unroll
    for (int k = 0; k < 3; ++k) atomicAdd(&cnum[rw * 3 + k], sCd[t * 3 + k] * cm);
    atomicAdd(&cden[rw], 1.0f);
  }
}

// ---------------------------------------------------------------------------
// A2V / V2A "edge" kernel over rows R = node*NCH + channel.
// Input row: [h_cur[i] | vnf[b,:,ch] | ||vcd||].
// mode 0 (A2V): efAcc = aggv[B,C,H], cdAcc = vcacc[B,3,C], cnt[b] += 1 (ch==0)
// mode 1 (V2A): efAcc = vefsum[N,H] (scaled 1/C), cdAcc[i] += -vcd*cm/C
// ---------------------------------------------------------------------------
__global__ void vedge_kernel(
    const float* __restrict__ hcur, const float* __restrict__ ccur,
    const float* __restrict__ vnf, const float* __restrict__ vc,
    const int* __restrict__ batch,
    const v16bf* __restrict__ w1f, const float* __restrict__ b1,
    const v16bf* __restrict__ w2f, const float* __restrict__ b2,
    const v16bf* __restrict__ cw1f, const float* __restrict__ cb1,
    const v16bf* __restrict__ cw2f,
    int mode, float efScale,
    float* __restrict__ efAcc, float* __restrict__ cdAcc, float* __restrict__ cnt) {
  __shared__ bf16_t A[TILE * 288];
  __shared__ bf16_t Mid[TILE * HDIM];
  __shared__ float sVcd[TILE * 3];
  __shared__ float sCm[TILE];
  __shared__ int   sI[TILE], sB[TILE], sCh[TILE], sBase[TILE];

  const int t = threadIdx.x, wave = t >> 5, lane = t & 31;
  const int r = t >> 2, s = t & 3;
  const int R = blockIdx.x * TILE + r;      // N*NCH multiple of TILE
  const int i = R / NCH, ch = R % NCH;
  const int b = batch[i];
  if (s == 0) {
    sI[r] = i; sB[r] = b; sCh[r] = ch;
    sBase[r] = (mode == 0) ? (b * NCH + ch) * HDIM : i * HDIM;
  }

  const float4* h4 = (const float4*)(hcur + (size_t)i * HDIM);
  __builtin_prefetch(h4, 0, 0);
  for (int q = s * 8; q < s * 8 + 8; ++q) {
    float4 a = h4[q];
    A[r * 288 + q * 4 + 0] = (bf16_t)a.x; A[r * 288 + q * 4 + 1] = (bf16_t)a.y;
    A[r * 288 + q * 4 + 2] = (bf16_t)a.z; A[r * 288 + q * 4 + 3] = (bf16_t)a.w;
  }
  for (int c = s * 32; c < s * 32 + 32; ++c)   // vnf[b, c, ch], stride NCH over features
    A[r * 288 + 128 + c] = (bf16_t)vnf[(size_t)(b * HDIM + c) * NCH + ch];
  if (s == 0) {
    float rad = 0.0f;
#pragma unroll
    for (int k = 0; k < 3; ++k) {
      float d = vc[(b * 3 + k) * NCH + ch] - ccur[i * 3 + k];
      sVcd[r * 3 + k] = d; rad += d * d;
    }
    A[r * 288 + 256] = (bf16_t)sqrtf(rad);
    for (int c = 257; c < 288; ++c) A[r * 288 + c] = (bf16_t)0.0f;
  }
  __syncthreads();

  gemm_tile64(A, 288, w1f, 9, b1, 1, Mid, HDIM, nullptr, nullptr, 0.0f, wave, lane);
  __syncthreads();
  gemm_tile64(Mid, HDIM, w2f, 4, b2, 1, A, HDIM, efAcc, sBase, efScale, wave, lane);
  __syncthreads();
  gemm_tile64(A, HDIM, cw1f, 4, cb1, 1, Mid, HDIM, nullptr, nullptr, 0.0f, wave, lane);
  __syncthreads();
  gemm_colvec64(Mid, HDIM, cw2f, 4, sCm, wave, lane);
  __syncthreads();

  if (t < TILE) {
    const float cm = sCm[t];
    if (mode == 0) {
      const int bb = sB[t], cc = sCh[t];
#pragma unroll
      for (int k = 0; k < 3; ++k)
        atomicAdd(&cdAcc[(bb * 3 + k) * NCH + cc], sVcd[t * 3 + k] * cm);
      if (cc == 0) atomicAdd(&cnt[bb], 1.0f);
    } else {
      const int ii = sI[t];
#pragma unroll
      for (int k = 0; k < 3; ++k)
        atomicAdd(&cdAcc[ii * 3 + k], -sVcd[t * 3 + k] * cm * (1.0f / NCH));
    }
  }
}

// ---------------------------------------------------------------------------
// Node MLP kernel (A2A and V2A node updates), 64 rows/block:
//   hout = hin + L2(silu(L1([hin | x2])))
//   coordOut = coordIn + cAdd (optionally / max(cDen,1))
// ---------------------------------------------------------------------------
__global__ void node_kernel(
    const float* __restrict__ hin, const float* __restrict__ x2,
    const v16bf* __restrict__ w1f, const float* __restrict__ b1,
    const v16bf* __restrict__ w2f, const float* __restrict__ b2,
    float* __restrict__ hout,
    const float* __restrict__ coordIn, const float* __restrict__ cAdd,
    const float* __restrict__ cDen, float* __restrict__ coordOut) {
  __shared__ bf16_t A[TILE * 256];
  __shared__ bf16_t Mid[TILE * HDIM];
  const int t = threadIdx.x, wave = t >> 5, lane = t & 31;
  const int row0 = blockIdx.x * TILE;       // N multiple of TILE
  const int r = t >> 2, s = t & 3;
  const int i = row0 + r;
  {
    const float4* h4 = (const float4*)(hin + (size_t)i * HDIM);
    const float4* x4 = (const float4*)(x2 + (size_t)i * HDIM);
    for (int q = s * 8; q < s * 8 + 8; ++q) {
      float4 a = h4[q], b = x4[q];
      A[r * 256 + q * 4 + 0] = (bf16_t)a.x; A[r * 256 + q * 4 + 1] = (bf16_t)a.y;
      A[r * 256 + q * 4 + 2] = (bf16_t)a.z; A[r * 256 + q * 4 + 3] = (bf16_t)a.w;
      A[r * 256 + 128 + q * 4 + 0] = (bf16_t)b.x; A[r * 256 + 128 + q * 4 + 1] = (bf16_t)b.y;
      A[r * 256 + 128 + q * 4 + 2] = (bf16_t)b.z; A[r * 256 + 128 + q * 4 + 3] = (bf16_t)b.w;
    }
  }
  __syncthreads();
  gemm_tile64(A, 256, w1f, 8, b1, 1, Mid, HDIM, nullptr, nullptr, 0.0f, wave, lane);
  __syncthreads();
  gemm_tile64_out(Mid, HDIM, w2f, 4, b2, hin, hout, row0, wave, lane);

  if (t < TILE) {
    const int ii = row0 + t;
#pragma unroll
    for (int k = 0; k < 3; ++k) {
      float add = cAdd[ii * 3 + k];
      if (cDen) add /= fmaxf(cDen[ii], 1.0f);
      coordOut[ii * 3 + k] = coordIn[ii * 3 + k] + add;
    }
  }
}

// ---------------------------------------------------------------------------
// Virtual-node update (B*C = 120 rows): plain VALU, negligible FLOPs.
//   vnf1[b,:,ch] = vnf + (silu([vnf | aggv/cnt] @ w1 + b1)) @ w2 + b2
//   vc1[b,k,ch]  = vc + vcacc / max(cnt,1)
// ---------------------------------------------------------------------------
__global__ void vnode_kernel(
    const float* __restrict__ vnf, const float* __restrict__ vc,
    const float* __restrict__ aggv, const float* __restrict__ aggcnt,
    const float* __restrict__ vcacc, const float* __restrict__ vccnt,
    const float* __restrict__ w1, const float* __restrict__ b1,
    const float* __restrict__ w2, const float* __restrict__ b2,
    float* __restrict__ vnf1, float* __restrict__ vc1) {
  const int bc = blockIdx.x;
  const int b = bc / NCH, ch = bc % NCH;
  __shared__ float x[256];
  __shared__ float hid[HDIM];
  const int f = threadIdx.x;
  const float cnt = fmaxf(aggcnt[b], 1.0f);
  x[f]        = vnf[(size_t)(b * HDIM + f) * NCH + ch];
  x[128 + f]  = aggv[(size_t)(b * NCH + ch) * HDIM + f] / cnt;
  __syncthreads();
  float acc = b1[f];
  for (int k = 0; k < 256; ++k) acc += x[k] * w1[k * HDIM + f];
  hid[f] = silu_f(acc);
  __syncthreads();
  float acc2 = b2[f];
  for (int k = 0; k < HDIM; ++k) acc2 += hid[k] * w2[k * HDIM + f];
  vnf1[(size_t)(b * HDIM + f) * NCH + ch] = x[f] + acc2;
  if (f < 3) {
    const float cc = fmaxf(vccnt[b], 1.0f);
    vc1[(b * 3 + f) * NCH + ch] = vc[(b * 3 + f) * NCH + ch] + vcacc[(b * 3 + f) * NCH + ch] / cc;
  }
}

// ---------------------------------------------------------------------------
extern "C" void kernel_launch(void* const* d_in, const int* in_sizes, int n_in,
                              void* d_out, int out_size, void* d_ws, size_t ws_size,
                              hipStream_t stream) {
  (void)n_in; (void)out_size; (void)ws_size;
  const float* h     = (const float*)d_in[0];
  const float* coord = (const float*)d_in[1];
  const float* vnf   = (const float*)d_in[2];
  const float* vc    = (const float*)d_in[3];
  const int*   ei    = (const int*)d_in[4];
  const int*   batch = (const int*)d_in[5];
  const int N = in_sizes[0] / HDIM;
  const int E = in_sizes[4] / 2;
  const int B = in_sizes[2] / (HDIM * NCH);

  // params: 3 layers (a2a, a2v, v2a) x 11 entries:
  // 0 e_w1, 1 e_b1, 2 e_w2, 3 e_b2, 4 n_w1, 5 n_b1, 6 n_w2, 7 n_b2, 8 c_w1, 9 c_b1, 10 c_w2
  auto PP = [&](int layer, int idx) -> const float* {
    return (const float*)d_in[6 + layer * 11 + idx];
  };

  // workspace bump allocator (256B aligned)
  char* base = (char*)d_ws;
  size_t off = 0;
  auto alloc = [&](size_t bytes) -> void* {
    off = (off + 255) & ~(size_t)255;
    void* p = base + off;
    off += bytes;
    return p;
  };

  v16bf *ew1f[3], *ew2f[3], *cw1f[3], *cw2f[3];
  for (int L = 0; L < 3; ++L) {
    ew1f[L] = (v16bf*)alloc((size_t)288 * 128 * 2);
    ew2f[L] = (v16bf*)alloc((size_t)128 * 128 * 2);
    cw1f[L] = (v16bf*)alloc((size_t)128 * 128 * 2);
    cw2f[L] = (v16bf*)alloc((size_t)128 * 16 * 2);
  }
  v16bf* nw1f_a2a = (v16bf*)alloc((size_t)256 * 128 * 2);
  v16bf* nw2f_a2a = (v16bf*)alloc((size_t)128 * 128 * 2);
  v16bf* nw1f_v2a = (v16bf*)alloc((size_t)256 * 128 * 2);
  v16bf* nw2f_v2a = (v16bf*)alloc((size_t)128 * 128 * 2);

  float* agg    = (float*)alloc((size_t)N * HDIM * 4);
  float* cnum   = (float*)alloc((size_t)N * 3 * 4);
  float* cden   = (float*)alloc((size_t)N * 4);
  float* h1     = (float*)alloc((size_t)N * HDIM * 4);
  float* coord1 = (float*)alloc((size_t)N * 3 * 4);
  float* vcacc  = (float*)alloc((size_t)B * 3 * NCH * 4);
  float* bcnt   = (float*)alloc((size_t)B * 4);
  float* aggv   = (float*)alloc((size_t)B * NCH * HDIM * 4);
  float* vnf1   = (float*)alloc((size_t)B * HDIM * NCH * 4);
  float* vc1    = (float*)alloc((size_t)B * 3 * NCH * 4);
  float* vefsum = (float*)alloc((size_t)N * HDIM * 4);
  float* cacc2  = (float*)alloc((size_t)N * 3 * 4);

  // weight fragment preparation (cheap; re-done every call for determinism)
  auto prep = [&](const float* w, v16bf* f, int K, int Kpad, int Nc, int Npad) {
    int total = (Kpad / 32) * (Npad / 16) * 512;
    prep_frag_kernel<<<(total + 255) / 256, 256, 0, stream>>>((const float*)w, (bf16_t*)f,
                                                              K, Kpad, Nc, Npad);
  };
  for (int L = 0; L < 3; ++L) {
    prep(PP(L, 0),  ew1f[L], 257, 288, 128, 128);
    prep(PP(L, 2),  ew2f[L], 128, 128, 128, 128);
    prep(PP(L, 8),  cw1f[L], 128, 128, 128, 128);
    prep(PP(L, 10), cw2f[L], 128, 128, 1,   16);
  }
  prep(PP(0, 4), nw1f_a2a, 256, 256, 128, 128);
  prep(PP(0, 6), nw2f_a2a, 128, 128, 128, 128);
  prep(PP(2, 4), nw1f_v2a, 256, 256, 128, 128);
  prep(PP(2, 6), nw2f_v2a, 128, 128, 128, 128);

  // zero accumulators (graph-capturable async memsets)
  (void)hipMemsetAsync(agg,    0, (size_t)N * HDIM * 4, stream);
  (void)hipMemsetAsync(cnum,   0, (size_t)N * 3 * 4, stream);
  (void)hipMemsetAsync(cden,   0, (size_t)N * 4, stream);
  (void)hipMemsetAsync(vcacc,  0, (size_t)B * 3 * NCH * 4, stream);
  (void)hipMemsetAsync(bcnt,   0, (size_t)B * 4, stream);
  (void)hipMemsetAsync(aggv,   0, (size_t)B * NCH * HDIM * 4, stream);
  (void)hipMemsetAsync(vefsum, 0, (size_t)N * HDIM * 4, stream);
  (void)hipMemsetAsync(cacc2,  0, (size_t)N * 3 * 4, stream);

  // ---------------- EGCL_A2A ----------------
  a2a_edge_kernel<<<E / TILE, 256, 0, stream>>>(
      h, coord, ei, E,
      ew1f[0], PP(0, 1), ew2f[0], PP(0, 3), cw1f[0], PP(0, 9), cw2f[0],
      agg, cnum, cden);
  node_kernel<<<N / TILE, 256, 0, stream>>>(
      h, agg, nw1f_a2a, PP(0, 5), nw2f_a2a, PP(0, 7),
      h1, coord, cnum, cden, coord1);

  // ---------------- EGCL_A2V ----------------
  vedge_kernel<<<(N * NCH) / TILE, 256, 0, stream>>>(
      h1, coord1, vnf, vc, batch,
      ew1f[1], PP(1, 1), ew2f[1], PP(1, 3), cw1f[1], PP(1, 9), cw2f[1],
      /*mode=*/0, /*efScale=*/1.0f, aggv, vcacc, bcnt);
  vnode_kernel<<<B * NCH, HDIM, 0, stream>>>(
      vnf, vc, aggv, bcnt, vcacc, bcnt,
      PP(1, 4), PP(1, 5), PP(1, 6), PP(1, 7), vnf1, vc1);

  // ---------------- EGCL_V2A ----------------
  vedge_kernel<<<(N * NCH) / TILE, 256, 0, stream>>>(
      h1, coord1, vnf1, vc1, batch,
      ew1f[2], PP(2, 1), ew2f[2], PP(2, 3), cw1f[2], PP(2, 9), cw2f[2],
      /*mode=*/1, /*efScale=*/1.0f / NCH, vefsum, cacc2, nullptr);

  float* outH = (float*)d_out;
  float* outC = outH + (size_t)N * HDIM;
  node_kernel<<<N / TILE, 256, 0, stream>>>(
      h1, vefsum, nw1f_v2a, PP(2, 5), nw2f_v2a, PP(2, 7),
      outH, coord1, cacc2, nullptr, outC);
}
